// MultiHeadAttention_49168785605387
// MI455X (gfx1250) — compile-verified, matrix operand order checked by reading, not run on hardware
//
#include <hip/hip_runtime.h>
#include <hip/hip_bf16.h>

// MultiHeadAttention forward, CDNA5 (gfx1250), wave32 + WMMA bf16.
// B=2, S=2048, D=1024, H=16, HD=64.

typedef __attribute__((ext_vector_type(16))) __bf16 v16bf;
typedef __attribute__((ext_vector_type(8)))  __bf16 v8bf;
typedef __attribute__((ext_vector_type(8)))  float  v8f;

#define BB 2
#define SS 2048
#define DD 1024
#define HH 16
#define HD 64

// Load one 16x32 bf16 A/B fragment (v16bf per lane).
// Layout (ISA 7.12.2): lane&15 selects row(A)/col(B); lane>>4 selects K-group.
// elements 0..7  <-> K = 8*kg + 0..7
// elements 8..15 <-> K = 8*kg + 16..23
static __device__ inline v16bf load_frag16(const __bf16* base, int kg) {
    union { v16bf v; v8bf h[2]; } u;
    u.h[0] = *(const v8bf*)(base + 8 * kg);
    u.h[1] = *(const v8bf*)(base + 16 + 8 * kg);
    return u.v;
}

static __device__ inline v8f wmma_bf16(v16bf a, v16bf b, v8f c) {
    return __builtin_amdgcn_wmma_f32_16x16x32_bf16(
        /*neg_a=*/false, a, /*neg_b=*/false, b,
        /*c_mod=*/(short)0, c, /*reuse_a=*/false, /*reuse_b=*/false);
}

// ---------------------------------------------------------------------------
// fp32 -> bf16 conversion (one-time repack of x and weights)
// ---------------------------------------------------------------------------
__global__ void cvt_f32_bf16(const float* __restrict__ in,
                             __bf16* __restrict__ out, int n) {
    int i = blockIdx.x * blockDim.x + threadIdx.x;
    if (i < n) out[i] = (__bf16)in[i];
}

// ---------------------------------------------------------------------------
// Y = A @ W.T   (A: [M,K] bf16 row-major, W: [N,K] bf16 row-major)
// One wave computes a 64x64 output tile: 4 A-frags x 4 B-frags -> 16 WMMAs
// per K=32 step (1:1 WMMA:b128-load ratio, 4x fragment reuse).
// MODE 0: store bf16 as [b, h, s, hd]          (Q, K buffers)
// MODE 1: store bf16 as [b, h, hd, s]          (V transposed)
// MODE 2: store fp32 as [m, n] with + bias[n]  (final output projection)
// ---------------------------------------------------------------------------
template <int MODE>
__global__ __launch_bounds__(32)
void gemm_bf16(const __bf16* __restrict__ A, const __bf16* __restrict__ W,
               void* __restrict__ out, const float* __restrict__ bias,
               int M, int N, int K) {
    const int lane = threadIdx.x;
    const int col  = lane & 15;   // M-row for A-frags, N-col for B-frags / C
    const int kg   = lane >> 4;
    const int m0   = blockIdx.y * 64;
    const int n0   = blockIdx.x * 64;

    const __bf16* arow[4];
    const __bf16* brow[4];
    #pragma unroll
    for (int i = 0; i < 4; ++i) {
        arow[i] = A + (size_t)(m0 + 16 * i + col) * K;
        brow[i] = W + (size_t)(n0 + 16 * i + col) * K;
    }

    v8f c[4][4] = {};

    #pragma unroll 1
    for (int k0 = 0; k0 < K; k0 += 32) {
        // prefetch next K slice into cache hierarchy (global_prefetch_b8)
        if (k0 + 32 < K) {
            __builtin_prefetch(arow[0] + k0 + 32, 0, 3);
            __builtin_prefetch(brow[0] + k0 + 32, 0, 3);
        }
        v16bf a[4], b[4];
        #pragma unroll
        for (int i = 0; i < 4; ++i) a[i] = load_frag16(arow[i] + k0, kg);
        #pragma unroll
        for (int j = 0; j < 4; ++j) b[j] = load_frag16(brow[j] + k0, kg);
        #pragma unroll
        for (int i = 0; i < 4; ++i)
            #pragma unroll
            for (int j = 0; j < 4; ++j)
                c[i][j] = wmma_bf16(a[i], b[j], c[i][j]);
    }

    // C layout: lane holds col (lane&15); element e holds row e + 8*(lane>>4)
    #pragma unroll
    for (int i = 0; i < 4; ++i) {
        #pragma unroll
        for (int j = 0; j < 4; ++j) {
            const int nc = n0 + 16 * j + col;
            #pragma unroll
            for (int e = 0; e < 8; ++e) {
                const int mr  = m0 + 16 * i + e + 8 * kg;
                const float v = c[i][j][e];
                if (MODE == 2) {
                    float* o = (float*)out;
                    o[(size_t)mr * N + nc] = v + bias[nc];
                } else {
                    __bf16* o = (__bf16*)out;
                    const int bb = mr >> 11, s  = mr & (SS - 1);   // SS = 2048
                    const int h  = nc >> 6,  hd = nc & (HD - 1);   // HD = 64
                    size_t idx;
                    if (MODE == 0) idx = (((size_t)(bb * HH + h)) * SS + s) * HD + hd;
                    else           idx = (((size_t)(bb * HH + h)) * HD + hd) * SS + s;
                    o[idx] = (__bf16)v;
                }
            }
        }
    }
}

// ---------------------------------------------------------------------------
// Flash-style causal attention: one wave per (b*h, 16-row Q tile).
// Streams K/V in 32-column chunks, online softmax, fp32 accumulators.
// ctx written as bf16 [b, s, h*HD] == [M, D] row-major for the final GEMM.
// ---------------------------------------------------------------------------
__global__ __launch_bounds__(32)
void attn_fa(const __bf16* __restrict__ qb, const __bf16* __restrict__ kb,
             const __bf16* __restrict__ vtb, __bf16* __restrict__ ctxb) {
    const int lane = threadIdx.x;
    const int col  = lane & 15;
    const int kg   = lane >> 4;
    const int qt   = blockIdx.x;       // q tile index (16 rows)
    const int bh   = blockIdx.y;       // b*H + h
    const int b    = bh >> 4;
    const int h    = bh & 15;

    const __bf16* qh = qb  + (size_t)bh * SS * HD;   // [s, hd]
    const __bf16* kh = kb  + (size_t)bh * SS * HD;   // [s, hd]
    const __bf16* vh = vtb + (size_t)bh * HD * SS;   // [hd, s]

    __shared__ float  s_tile[16][32];
    __shared__ __bf16 w_tile[16][32];
    __shared__ float  s_scale[16];

    // Q tile fragments, K-dim = HD = 64 -> two 16x32 fragments (loaded once)
    const v16bf aq0 = load_frag16(qh + (size_t)(qt * 16 + col) * HD +  0, kg);
    const v16bf aq1 = load_frag16(qh + (size_t)(qt * 16 + col) * HD + 32, kg);

    v8f acc[4] = {};                // ctx accumulator, 16 x 64
    float m_r = -3.0e38f;           // running row max  (owned by lanes 0..15)
    float l_r = 0.0f;               // running row sum

    const int nchunks = (qt * 16 + 47) / 32;   // causal bound, 32-col chunks
    #pragma unroll 1
    for (int ck = 0; ck < nchunks; ++ck) {
        const int kbase = ck * 32;

        // ---- scores: two 16x16 fragments covering 32 key columns ----
        #pragma unroll
        for (int c0 = 0; c0 < 32; c0 += 16) {
            const __bf16* krow = kh + (size_t)(kbase + c0 + col) * HD;
            v16bf bk0 = load_frag16(krow +  0, kg);
            v16bf bk1 = load_frag16(krow + 32, kg);
            v8f sc = {};
            sc = wmma_bf16(aq0, bk0, sc);
            sc = wmma_bf16(aq1, bk1, sc);
            #pragma unroll
            for (int j = 0; j < 8; ++j) {
                const int r  = j + 8 * kg;
                const int gr = qt * 16 + r;
                const int gc = kbase + c0 + col;
                float v = sc[j];
                if (gc > gr) v = -3.0e38f;      // causal mask
                s_tile[r][c0 + col] = v;
            }
        }
        __syncthreads();

        // ---- online softmax over this 32-wide chunk (lane r owns row r) ----
        if (lane < 16) {
            float mx = m_r;
            for (int cc = 0; cc < 32; ++cc)
                mx = fmaxf(mx, s_tile[lane][cc] * 0.125f);   // 1/sqrt(64)
            float sum = 0.0f;
            for (int cc = 0; cc < 32; ++cc) {
                float p = __expf(s_tile[lane][cc] * 0.125f - mx);
                sum += p;
                w_tile[lane][cc] = (__bf16)p;
            }
            float sc = __expf(m_r - mx);
            l_r = l_r * sc + sum;
            m_r = mx;
            s_scale[lane] = sc;
        }
        __syncthreads();

        // ---- rescale accumulators by per-row correction factor ----
        float rs[8];
        #pragma unroll
        for (int j = 0; j < 8; ++j) rs[j] = s_scale[j + 8 * kg];
        #pragma unroll
        for (int f = 0; f < 4; ++f)
            #pragma unroll
            for (int j = 0; j < 8; ++j)
                acc[f][j] *= rs[j];

        // ---- ctx += P @ V : A-frag from LDS, B-frags lane-contiguous in v^T
        v16bf aw = load_frag16(&w_tile[col][0], kg);
        #pragma unroll
        for (int f = 0; f < 4; ++f) {
            v16bf bv = load_frag16(vh + (size_t)(f * 16 + col) * SS + kbase, kg);
            acc[f] = wmma_bf16(aw, bv, acc[f]);
        }
        __syncthreads();   // before next chunk overwrites s_tile / w_tile
    }

    // ---- normalize by row sum and store ctx as bf16 [m, h*HD + d] ----
    if (lane < 16) s_scale[lane] = 1.0f / l_r;
    __syncthreads();
    float inv[8];
    #pragma unroll
    for (int j = 0; j < 8; ++j) inv[j] = s_scale[j + 8 * kg];
    #pragma unroll
    for (int f = 0; f < 4; ++f) {
        #pragma unroll
        for (int j = 0; j < 8; ++j) {
            const int mr = b * SS + qt * 16 + j + 8 * kg;
            const int nc = h * HD + f * 16 + col;
            ctxb[(size_t)mr * DD + nc] = (__bf16)(acc[f][j] * inv[j]);
        }
    }
}

// ---------------------------------------------------------------------------
extern "C" void kernel_launch(void* const* d_in, const int* in_sizes, int n_in,
                              void* d_out, int out_size, void* d_ws, size_t ws_size,
                              hipStream_t stream) {
    (void)in_sizes; (void)n_in; (void)out_size; (void)ws_size;

    const float* x  = (const float*)d_in[0];
    const float* Wq = (const float*)d_in[1];
    const float* Wk = (const float*)d_in[2];
    const float* Wv = (const float*)d_in[3];
    const float* Wo = (const float*)d_in[4];
    const float* bo = (const float*)d_in[5];

    const size_t nX = (size_t)BB * SS * DD;   // 4,194,304 elems
    const size_t nW = (size_t)DD * DD;        // 1,048,576 elems

    char* ws = (char*)d_ws;
    size_t off = 0;
    __bf16* x_bf  = (__bf16*)(ws + off); off += nX * 2;
    __bf16* wq_bf = (__bf16*)(ws + off); off += nW * 2;
    __bf16* wk_bf = (__bf16*)(ws + off); off += nW * 2;
    __bf16* wv_bf = (__bf16*)(ws + off); off += nW * 2;
    __bf16* wo_bf = (__bf16*)(ws + off); off += nW * 2;
    __bf16* qbuf  = (__bf16*)(ws + off); off += nX * 2;   // [b,h,s,hd]
    __bf16* kbuf  = (__bf16*)(ws + off); off += nX * 2;   // [b,h,s,hd]
    __bf16* vtbuf = (__bf16*)(ws + off); off += nX * 2;   // [b,h,hd,s]
    __bf16* ctxbf = (__bf16*)(ws + off); off += nX * 2;   // [m, d]

    // 1) fp32 -> bf16 repack
    {
        int t = 256;
        cvt_f32_bf16<<<(int)((nX + t - 1) / t), t, 0, stream>>>(x,  x_bf,  (int)nX);
        cvt_f32_bf16<<<(int)((nW + t - 1) / t), t, 0, stream>>>(Wq, wq_bf, (int)nW);
        cvt_f32_bf16<<<(int)((nW + t - 1) / t), t, 0, stream>>>(Wk, wk_bf, (int)nW);
        cvt_f32_bf16<<<(int)((nW + t - 1) / t), t, 0, stream>>>(Wv, wv_bf, (int)nW);
        cvt_f32_bf16<<<(int)((nW + t - 1) / t), t, 0, stream>>>(Wo, wo_bf, (int)nW);
    }

    // 2) Q/K/V projections (M=4096, N=K=1024), 64x64 tile per wave
    dim3 gblk(32);
    dim3 ggrd(DD / 64, (BB * SS) / 64);
    gemm_bf16<0><<<ggrd, gblk, 0, stream>>>(x_bf, wq_bf, qbuf,  nullptr, BB * SS, DD, DD);
    gemm_bf16<0><<<ggrd, gblk, 0, stream>>>(x_bf, wk_bf, kbuf,  nullptr, BB * SS, DD, DD);
    gemm_bf16<1><<<ggrd, gblk, 0, stream>>>(x_bf, wv_bf, vtbuf, nullptr, BB * SS, DD, DD);

    // 3) causal flash attention
    attn_fa<<<dim3(SS / 16, BB * HH), 32, 0, stream>>>(qbuf, kbuf, vtbuf, ctxbf);

    // 4) output projection + bias (fp32 result)
    gemm_bf16<2><<<ggrd, gblk, 0, stream>>>(ctxbf, wo_bf, d_out, bo, BB * SS, DD, DD);
}
